// PredictiveStreamingBlock_45612552683662
// MI455X (gfx1250) — compile-verified, matrix operand order checked by reading, not run on hardware
//
#include <hip/hip_runtime.h>
#include <hip/hip_bf16.h>
#include <math.h>

#define D_MODEL 1024
#define D_FF    4096
#define NEXP    8
#define TOKENS  8192            // 4 * 2048
#define LN_EPS  1e-5f

#define BM 128
#define BN 128
#define BK 32
#define LDT 40                  // padded K-stride (elements) for LDS tiles
#define TT 32                   // transpose tile

typedef unsigned short u16;
typedef __attribute__((ext_vector_type(16))) __bf16 v16bf;
typedef __attribute__((ext_vector_type(8)))  float  v8f;

union Frag16 { v16bf v; uint4 q[2]; };

__device__ __forceinline__ u16 f2bf(float f) {
  unsigned u = __float_as_uint(f);
  u += 0x7FFFu + ((u >> 16) & 1u);          // round-to-nearest-even
  return (u16)(u >> 16);
}

// gelu(tanh approx): 0.5*v*(1+tanh(a)) == v - v/(1+exp(2a)),  a = sqrt(2/pi)*(v+0.044715 v^3)
__device__ __forceinline__ float gelu_tanh(float v) {
  float u = 1.5957691216057308f * (v + 0.044715f * v * v * v);   // 2*a
  return v - v * __frcp_rn(1.0f + __expf(u));
}

// ---------------- init: zero per-expert counters ----------------
__global__ void init_counts_kernel(int* __restrict__ counts) {
  if (threadIdx.x < NEXP) counts[threadIdx.x] = 0;
}

// ---------------- fp32 -> bf16 (8 elements / thread, same layout) ----------------
__global__ void cvt_bf16_kernel(const float* __restrict__ src, u16* __restrict__ dst) {
  size_t i = (size_t)(blockIdx.x * blockDim.x + threadIdx.x) * 8;
  float4 a = *(const float4*)(src + i);
  float4 b = *(const float4*)(src + i + 4);
  uint4 o;
  o.x = (unsigned)f2bf(a.x) | ((unsigned)f2bf(a.y) << 16);
  o.y = (unsigned)f2bf(a.z) | ((unsigned)f2bf(a.w) << 16);
  o.z = (unsigned)f2bf(b.x) | ((unsigned)f2bf(b.y) << 16);
  o.w = (unsigned)f2bf(b.z) | ((unsigned)f2bf(b.w) << 16);
  *(uint4*)(dst + i) = o;
}

// ---------------- fp32 [E][K][N] -> bf16 [E][N][K] tiled transpose ----------------
__global__ void cvt_transpose_kernel(const float* __restrict__ src, u16* __restrict__ dst,
                                     int K, int N) {
  int e = blockIdx.z;
  const float* s = src + (size_t)e * K * N;
  u16* d = dst + (size_t)e * K * N;
  int kBase = blockIdx.y * TT, nBase = blockIdx.x * TT;
  __shared__ __align__(8) u16 tile[TT][TT + 4];
  int r = threadIdx.x >> 3;            // 0..31
  int c = (threadIdx.x & 7) * 4;       // 0,4,...,28
  float4 v = *(const float4*)(s + (size_t)(kBase + r) * N + nBase + c);
  uint2 p;
  p.x = (unsigned)f2bf(v.x) | ((unsigned)f2bf(v.y) << 16);
  p.y = (unsigned)f2bf(v.z) | ((unsigned)f2bf(v.w) << 16);
  *(uint2*)&tile[r][c] = p;
  __syncthreads();
  uint2 o;
  u16* op = (u16*)&o;
  op[0] = tile[c + 0][r];
  op[1] = tile[c + 1][r];
  op[2] = tile[c + 2][r];
  op[3] = tile[c + 3][r];
  *(uint2*)(d + (size_t)(nBase + r) * K + kBase + c) = o;
}

// ---------------- router: logits -> softmax -> top-2, comb + rank-tagged lists ----------------
__global__ void router_kernel(const float* __restrict__ x, const float* __restrict__ rw,
                              const float* __restrict__ rb, float* __restrict__ comb,
                              int* __restrict__ counts, int* __restrict__ lists) {
  int wave = threadIdx.x >> 5, lane = threadIdx.x & 31;
  int t = blockIdx.x * (blockDim.x >> 5) + wave;
  const float* xr = x + (size_t)t * D_MODEL;
  float acc[NEXP];
#pragma unroll
  for (int e = 0; e < NEXP; e++) acc[e] = 0.f;
  for (int d = lane; d < D_MODEL; d += 32) {
    float xv = xr[d];
    const float4* r4 = (const float4*)(rw + (size_t)d * NEXP);
    float4 r0 = r4[0], r1 = r4[1];
    acc[0] = fmaf(xv, r0.x, acc[0]); acc[1] = fmaf(xv, r0.y, acc[1]);
    acc[2] = fmaf(xv, r0.z, acc[2]); acc[3] = fmaf(xv, r0.w, acc[3]);
    acc[4] = fmaf(xv, r1.x, acc[4]); acc[5] = fmaf(xv, r1.y, acc[5]);
    acc[6] = fmaf(xv, r1.z, acc[6]); acc[7] = fmaf(xv, r1.w, acc[7]);
  }
#pragma unroll
  for (int e = 0; e < NEXP; e++)
#pragma unroll
    for (int m = 16; m > 0; m >>= 1) acc[e] += __shfl_xor(acc[e], m, 32);
  if (lane == 0) {
    float logit[NEXP], p[NEXP];
    float mx = -1e30f;
#pragma unroll
    for (int e = 0; e < NEXP; e++) { logit[e] = acc[e] + rb[e]; mx = fmaxf(mx, logit[e]); }
    float sum = 0.f;
#pragma unroll
    for (int e = 0; e < NEXP; e++) { p[e] = __expf(logit[e] - mx); sum += p[e]; }
    float inv = 1.f / sum;
    int i0 = 0; float v0 = -1.f;
#pragma unroll
    for (int e = 0; e < NEXP; e++) if (p[e] > v0) { v0 = p[e]; i0 = e; }
    int i1 = (i0 == 0) ? 1 : 0; float v1 = -1.f;
#pragma unroll
    for (int e = 0; e < NEXP; e++) if (e != i0 && p[e] > v1) { v1 = p[e]; i1 = e; }
    float* cr = comb + (size_t)t * NEXP;
#pragma unroll
    for (int e = 0; e < NEXP; e++) cr[e] = 0.f;
    cr[i0] = v0 * inv;
    cr[i1] = v1 * inv;
    int p0 = atomicAdd(&counts[i0], 1); lists[i0 * TOKENS + p0] = t;               // rank 0
    int p1 = atomicAdd(&counts[i1], 1); lists[i1 * TOKENS + p1] = t | (1 << 30);   // rank 1
  }
}

// ---------------- prefix sum of counts -> slot offsets; emit selected mask ----------------
__global__ void offsets_kernel(const int* __restrict__ counts, int* __restrict__ offsets,
                               float* __restrict__ out_sel) {
  if (threadIdx.x == 0) {
    int off = 0;
    for (int e = 0; e < NEXP; e++) {
      offsets[e] = off;
      off += counts[e];
      out_sel[e] = counts[e] > 0 ? 1.0f : 0.0f;
    }
  }
}

// ---------------- shared WMMA inner step: 2x4 tiles of 16x16, K=32 ----------------
__device__ __forceinline__ void wmma_step(const u16* lA, const u16* lB, int lane,
                                          int mW, int nW, v8f acc[2][4]) {
  int kh = (lane >= 16) ? 8 : 0;     // K half select per ISA 16-bit A/B lane layout
  int rl = lane & 15;
  Frag16 fa[2], fb[4];
#pragma unroll
  for (int i = 0; i < 2; i++) {
    const u16* p = lA + (mW + i * 16 + rl) * LDT + kh;
    fa[i].q[0] = *(const uint4*)p;
    fa[i].q[1] = *(const uint4*)(p + 16);
  }
#pragma unroll
  for (int j = 0; j < 4; j++) {
    const u16* p = lB + (nW + j * 16 + rl) * LDT + kh;
    fb[j].q[0] = *(const uint4*)p;
    fb[j].q[1] = *(const uint4*)(p + 16);
  }
#pragma unroll
  for (int i = 0; i < 2; i++)
#pragma unroll
    for (int j = 0; j < 4; j++)
      acc[i][j] = __builtin_amdgcn_wmma_f32_16x16x32_bf16(
          false, fa[i].v, false, fb[j].v, (short)0, acc[i][j], false, false);
}

// ---------------- FFN1: H = gelu(Xe @ W1[e] + b1[e]) for dispatched rows ----------------
__global__ __launch_bounds__(256) void ffn1_kernel(
    const u16* __restrict__ xb, const u16* __restrict__ w1t, const float* __restrict__ b1,
    const int* __restrict__ counts, const int* __restrict__ offsets,
    const int* __restrict__ lists, u16* __restrict__ Hb) {
  int e = blockIdx.z;
  int cnt = counts[e];
  int mBase = blockIdx.y * BM;
  if (mBase >= cnt) return;
  int nBase = blockIdx.x * BN;

  __shared__ __align__(16) u16 lA[BM * LDT];
  __shared__ __align__(16) u16 lB[BN * LDT];
  __shared__ int tok[BM];

  int tid = threadIdx.x;
  if (tid < BM) {
    int r = mBase + tid;
    tok[tid] = lists[e * TOKENS + (r < cnt ? r : cnt - 1)] & 0x3FFFFFFF;
  }
  __syncthreads();

  int wave = tid >> 5, lane = tid & 31;
  int mW = (wave >> 1) * 32, nW = (wave & 1) * 64;
  v8f acc[2][4];
#pragma unroll
  for (int i = 0; i < 2; i++)
#pragma unroll
    for (int j = 0; j < 4; j++) acc[i][j] = (v8f){};

  int arow = tid >> 1, acol = (tid & 1) * 16;   // also B row/col: both tiles K-contiguous
  const u16* aPtr = xb + (size_t)tok[arow] * D_MODEL + acol;
  const u16* bPtr = w1t + (size_t)e * D_FF * D_MODEL + (size_t)(nBase + arow) * D_MODEL + acol;
  u16* lAp = lA + arow * LDT + acol;
  u16* lBp = lB + arow * LDT + acol;

  // software pipeline: regs hold tile k0, global loads for k0+BK overlap the WMMAs
  uint4 a0 = *(const uint4*)(aPtr);
  uint4 a1 = *(const uint4*)(aPtr + 8);
  uint4 b0 = *(const uint4*)(bPtr);
  uint4 b1v = *(const uint4*)(bPtr + 8);
  for (int k0 = 0; k0 < D_MODEL; k0 += BK) {
    __syncthreads();
    *(uint4*)(lAp) = a0;
    *(uint4*)(lAp + 8) = a1;
    *(uint4*)(lBp) = b0;
    *(uint4*)(lBp + 8) = b1v;
    __syncthreads();
    int kn = (k0 + BK < D_MODEL) ? (k0 + BK) : 0;
    a0 = *(const uint4*)(aPtr + kn);
    a1 = *(const uint4*)(aPtr + kn + 8);
    b0 = *(const uint4*)(bPtr + kn);
    b1v = *(const uint4*)(bPtr + kn + 8);
    wmma_step(lA, lB, lane, mW, nW, acc);
  }

  int slotBase = offsets[e];
  int nl = lane & 15, mh = (lane >> 4) * 8;
#pragma unroll
  for (int j = 0; j < 4; j++) {
    int n = nBase + nW + j * 16 + nl;
    float bias = b1[(size_t)e * D_FF + n];
#pragma unroll
    for (int i = 0; i < 2; i++) {
#pragma unroll
      for (int r = 0; r < 8; r++) {
        int m = mW + i * 16 + mh + r;
        if (mBase + m < cnt) {
          float v = acc[i][j][r] + bias;
          Hb[(size_t)(slotBase + mBase + m) * D_FF + n] = f2bf(gelu_tanh(v));
        }
      }
    }
  }
}

// ---------------- FFN2: part[rank] = comb * (H @ W2[e] + b2[e])  (no atomics) ----------------
__global__ __launch_bounds__(256) void ffn2_kernel(
    const u16* __restrict__ Hb, const u16* __restrict__ w2t, const float* __restrict__ b2,
    const float* __restrict__ comb, const int* __restrict__ counts,
    const int* __restrict__ offsets, const int* __restrict__ lists,
    float* __restrict__ part0, float* __restrict__ part1) {
  int e = blockIdx.z;
  int cnt = counts[e];
  int mBase = blockIdx.y * BM;
  if (mBase >= cnt) return;
  int nBase = blockIdx.x * BN;
  int slotBase = offsets[e];

  __shared__ __align__(16) u16 lA[BM * LDT];
  __shared__ __align__(16) u16 lB[BN * LDT];

  int tid = threadIdx.x;
  int wave = tid >> 5, lane = tid & 31;
  int mW = (wave >> 1) * 32, nW = (wave & 1) * 64;
  v8f acc[2][4];
#pragma unroll
  for (int i = 0; i < 2; i++)
#pragma unroll
    for (int j = 0; j < 4; j++) acc[i][j] = (v8f){};

  int arow = tid >> 1, acol = (tid & 1) * 16;
  int ar = mBase + arow; if (ar >= cnt) ar = cnt - 1;
  const u16* aPtr = Hb + (size_t)(slotBase + ar) * D_FF + acol;
  const u16* bPtr = w2t + (size_t)e * D_FF * D_MODEL + (size_t)(nBase + arow) * D_FF + acol;
  u16* lAp = lA + arow * LDT + acol;
  u16* lBp = lB + arow * LDT + acol;

  uint4 a0 = *(const uint4*)(aPtr);
  uint4 a1 = *(const uint4*)(aPtr + 8);
  uint4 b0 = *(const uint4*)(bPtr);
  uint4 b1v = *(const uint4*)(bPtr + 8);
  for (int k0 = 0; k0 < D_FF; k0 += BK) {
    __syncthreads();
    *(uint4*)(lAp) = a0;
    *(uint4*)(lAp + 8) = a1;
    *(uint4*)(lBp) = b0;
    *(uint4*)(lBp + 8) = b1v;
    __syncthreads();
    int kn = (k0 + BK < D_FF) ? (k0 + BK) : 0;
    a0 = *(const uint4*)(aPtr + kn);
    a1 = *(const uint4*)(aPtr + kn + 8);
    b0 = *(const uint4*)(bPtr + kn);
    b1v = *(const uint4*)(bPtr + kn + 8);
    wmma_step(lA, lB, lane, mW, nW, acc);
  }

  int nl = lane & 15, mh = (lane >> 4) * 8;
  float biasv[4];
#pragma unroll
  for (int j = 0; j < 4; j++) biasv[j] = b2[(size_t)e * D_MODEL + nBase + nW + j * 16 + nl];
#pragma unroll
  for (int i = 0; i < 2; i++) {
#pragma unroll
    for (int r = 0; r < 8; r++) {
      int row = mBase + mW + i * 16 + mh + r;
      if (row < cnt) {
        int entry = lists[e * TOKENS + row];
        int t = entry & 0x3FFFFFFF;
        float s = comb[(size_t)t * NEXP + e];
        float* dst = ((entry >> 30) ? part1 : part0) + (size_t)t * D_MODEL;
#pragma unroll
        for (int j = 0; j < 4; j++) {
          int n = nBase + nW + j * 16 + nl;
          dst[n] = s * (acc[i][j][r] + biasv[j]);
        }
      }
    }
  }
}

// ---------------- residual + LayerNorm ----------------
__global__ void ln_kernel(const float* __restrict__ x, const float* __restrict__ p0,
                          const float* __restrict__ p1, const float* __restrict__ gamma,
                          const float* __restrict__ beta, float* __restrict__ z) {
  int wave = threadIdx.x >> 5, lane = threadIdx.x & 31;
  int t = blockIdx.x * (blockDim.x >> 5) + wave;
  const float* xr = x + (size_t)t * D_MODEL;
  const float* a0 = p0 + (size_t)t * D_MODEL;
  const float* a1 = p1 + (size_t)t * D_MODEL;
  float s = 0.f, s2 = 0.f;
  for (int d = lane; d < D_MODEL; d += 32) {
    float y = xr[d] + a0[d] + a1[d];
    s += y; s2 += y * y;
  }
#pragma unroll
  for (int m = 16; m > 0; m >>= 1) { s += __shfl_xor(s, m, 32); s2 += __shfl_xor(s2, m, 32); }
  float mu = s * (1.0f / D_MODEL);
  float var = s2 * (1.0f / D_MODEL) - mu * mu;
  float rs = rsqrtf(var + LN_EPS);
  float* zr = z + (size_t)t * D_MODEL;
  for (int d = lane; d < D_MODEL; d += 32) {
    float y = xr[d] + a0[d] + a1[d];
    zr[d] = (y - mu) * rs * gamma[d] + beta[d];
  }
}

extern "C" void kernel_launch(void* const* d_in, const int* in_sizes, int n_in,
                              void* d_out, int out_size, void* d_ws, size_t ws_size,
                              hipStream_t stream) {
  const float* x  = (const float*)d_in[0];
  const float* rw = (const float*)d_in[1];
  const float* rb = (const float*)d_in[2];
  const float* w1 = (const float*)d_in[3];
  const float* b1 = (const float*)d_in[4];
  const float* w2 = (const float*)d_in[5];
  const float* b2 = (const float*)d_in[6];
  const float* gamma = (const float*)d_in[7];
  const float* beta  = (const float*)d_in[8];
  float* out = (float*)d_out;

  // workspace layout (all 256B-aligned)
  char* ws = (char*)d_ws;
  size_t off = 0;
  u16* xb   = (u16*)(ws + off); off += (size_t)TOKENS * D_MODEL * 2;          // 16 MB
  u16* w1t  = (u16*)(ws + off); off += (size_t)NEXP * D_MODEL * D_FF * 2;     // 64 MB (transposed)
  u16* w2t  = (u16*)(ws + off); off += (size_t)NEXP * D_FF * D_MODEL * 2;     // 64 MB (transposed)
  u16* Hb   = (u16*)(ws + off); off += (size_t)2 * TOKENS * D_FF * 2;         // 128 MB
  float* part0 = (float*)(ws + off); off += (size_t)TOKENS * D_MODEL * 4;     // 32 MB
  float* part1 = (float*)(ws + off); off += (size_t)TOKENS * D_MODEL * 4;     // 32 MB
  float* comb  = (float*)(ws + off); off += (size_t)TOKENS * NEXP * 4;        // 256 KB
  int* lists   = (int*)(ws + off);   off += (size_t)NEXP * TOKENS * 4;        // 256 KB
  int* counts  = (int*)(ws + off);   off += 256;
  int* offsets = (int*)(ws + off);   off += 256;
  (void)ws_size; (void)in_sizes; (void)n_in; (void)out_size;

  // 1. zero dispatch counters
  init_counts_kernel<<<1, 32, 0, stream>>>(counts);
  // 2. fp32 -> bf16 conversions; weights transposed to [N][K] so GEMM tiles are K-contiguous
  cvt_bf16_kernel<<<(TOKENS * D_MODEL / 8) / 256, 256, 0, stream>>>(x, xb);
  cvt_transpose_kernel<<<dim3(D_FF / TT, D_MODEL / TT, NEXP), 256, 0, stream>>>(
      w1, w1t, D_MODEL, D_FF);
  cvt_transpose_kernel<<<dim3(D_MODEL / TT, D_FF / TT, NEXP), 256, 0, stream>>>(
      w2, w2t, D_FF, D_MODEL);
  // 3. routing + rank-tagged dispatch
  router_kernel<<<TOKENS / 8, 256, 0, stream>>>(x, rw, rb, comb, counts, lists);
  // 4. slot offsets + selected-expert mask (output tail)
  offsets_kernel<<<1, 32, 0, stream>>>(counts, offsets, out + (size_t)TOKENS * D_MODEL);
  // 5. expert FFN layer 1 (gathered GEMM + gelu)
  ffn1_kernel<<<dim3(D_FF / BN, TOKENS / BM, NEXP), 256, 0, stream>>>(
      xb, w1t, b1, counts, offsets, lists, Hb);
  // 6. expert FFN layer 2 -> rank-partitioned partial outputs (no atomics)
  ffn2_kernel<<<dim3(D_MODEL / BN, TOKENS / BM, NEXP), 256, 0, stream>>>(
      Hb, w2t, b2, comb, counts, offsets, lists, part0, part1);
  // 7. residual + LayerNorm -> z
  ln_kernel<<<TOKENS / 8, 256, 0, stream>>>(x, part0, part1, gamma, beta, out);
}
